// CONV_minimal_add_partial_558345748884
// MI455X (gfx1250) — compile-verified, matrix operand order checked by reading, not run on hardware
//
#include <hip/hip_runtime.h>
#include <hip/hip_bf16.h>

typedef _Float16 f16;
typedef __attribute__((ext_vector_type(16))) _Float16 v16h;
typedef __attribute__((ext_vector_type(8)))  _Float16 v8h;
typedef __attribute__((ext_vector_type(8)))  float    v8f;

#define BATCH 16384
#define EPS 1e-5f
#define H1_POS   196           // 14x14 positions per image
#define H1_CH    8             // 6 valid channels + 2 zero pad (NHWC8)
#define H1_STRIDE (H1_POS * H1_CH)   // 1568 halves per image

static __device__ inline v8f wmma_f16(v16h a, v16h b, v8f c) {
    return __builtin_amdgcn_wmma_f32_16x16x32_f16(false, a, false, b, (short)0, c, false, false);
}

static __device__ inline float clip01(float v) {
    return fminf(fmaxf(v, 0.0f), 1.0f);
}

// chunk index c8 = ky*5+kx (c8<25 valid) -> h1 offset (ky*14+kx)*8 = (9*ky+c8)*8
static __device__ inline int conv2_chunk_off(int c8) {
    int ky  = (c8 * 205) >> 10;          // c8 / 5 for c8 < 32
    int off = (9 * ky + c8) * 8;
    return (c8 < 25) ? off : 0;          // pad chunks alias offset 0 (B pack is zero)
}

// ---------------------------------------------------------------------------
// stats layout (floats) in ws:
//  [0..5]    bn1 sum      [6..11]   bn1 sumsq
//  [12..17]  bn1 scale    [18..23]  bn1 shift
//  [24..39]  bn2 sum      [40..55]  bn2 sumsq
//  [56..71]  bn2 scale    [72..87]  bn2 shift
//  [88..97]  bn3 sum      [98..107] bn3 sumsq
//  [108..117] bn3 scale   [118..127] bn3 shift
// ---------------------------------------------------------------------------

__global__ void k_zero_stats(float* stats) {
    int t = threadIdx.x;
    if (t < 128) stats[t] = 0.0f;
}

// Pack weight matrix W[N][K] (row-major f32) into per-lane WMMA B fragments.
// pack[((kstep*ntiles + nt)*32 + lane)*16 + e], zero padded outside K,N.
__global__ void k_prep_pack(const float* __restrict__ W, f16* __restrict__ pack,
                            int K, int N, int ksteps, int ntiles) {
    int t = blockIdx.x * blockDim.x + threadIdx.x;
    int total = ksteps * ntiles * 512;
    if (t >= total) return;
    int e    = t & 15;
    int lane = (t >> 4) & 31;
    int sn   = t >> 9;
    int nt   = sn % ntiles;
    int kstep = sn / ntiles;
    int n  = nt * 16 + (lane & 15);
    int hs = lane >> 4;
    int k  = kstep * 32 + (e < 8 ? hs * 8 + e : 16 + hs * 8 + (e - 8));
    float v = 0.0f;
    if (k < K && n < N) v = W[(size_t)n * K + k];
    pack[t] = (f16)v;
}

// conv2 B-pack with the NHWC8 k-order: k' = (ky*5+kx)*8 + ch, K' = 256.
__global__ void k_prep_pack_conv2(const float* __restrict__ w2, f16* __restrict__ pack) {
    int t = blockIdx.x * blockDim.x + threadIdx.x;
    if (t >= 8 * 512) return;
    int e     = t & 15;
    int lane  = (t >> 4) & 31;
    int kstep = t >> 9;
    int n  = lane & 15;
    int hs = lane >> 4;
    int k  = kstep * 32 + (e < 8 ? hs * 8 + e : 16 + hs * 8 + (e - 8));
    int c8 = k >> 3;                      // position chunk (ky*5+kx)
    int ch = k & 7;                       // input channel
    float v = 0.0f;
    if (c8 < 25 && ch < 6) {
        int ky = c8 / 5, kx = c8 % 5;
        v = w2[(size_t)n * 150 + ch * 25 + ky * 5 + kx];
    }
    pack[t] = (f16)v;
}

// Effective 6x6 stride-2 filter for fused conv1+avgpool:
// W2[c][r][cc] = 0.25 * sum_{dy,dx in {0,1}} w1[c][r-dy][cc-dx]  (valid taps)
__global__ void k_prep_w2eff(const float* __restrict__ w1, float* __restrict__ w2eff) {
    int t = threadIdx.x;
    if (t >= 216) return;
    int c  = t / 36;
    int rc = t % 36;
    int r  = rc / 6, cc = rc % 6;
    float s = 0.0f;
    #pragma unroll
    for (int dy = 0; dy < 2; ++dy) {
        #pragma unroll
        for (int dx = 0; dx < 2; ++dx) {
            int ky = r - dy, kx = cc - dx;
            if (ky >= 0 && ky < 5 && kx >= 0 && kx < 5)
                s += w1[c * 25 + ky * 5 + kx];
        }
    }
    w2eff[t] = 0.25f * s;
}

// Fused conv1(5x5,pad2)+avgpool2 as a 6x6 stride-2 correlation. One thread
// loads a 6x6 patch and produces all 6 channels (NHWC8 output); bn1 sums.
__global__ void k_conv1_pool(const float* __restrict__ x, const float* __restrict__ w2eff,
                             f16* __restrict__ h1, float* __restrict__ stats) {
    __shared__ float s[12];
    int tid = threadIdx.x;
    if (tid < 12) s[tid] = 0.0f;
    __syncthreads();

    long idx = (long)blockIdx.x * blockDim.x + tid;
    const long total = (long)BATCH * H1_POS;
    if (idx < total) {
        int b  = (int)(idx / H1_POS);
        int pp = (int)(idx % H1_POS);
        int py = pp / 14, px = pp % 14;
        const float* xb = x + (size_t)b * 784;
        float in[36];
        #pragma unroll
        for (int r = 0; r < 6; ++r) {
            int iy = 2 * py - 2 + r;
            #pragma unroll
            for (int j = 0; j < 6; ++j) {
                int ix = 2 * px - 2 + j;
                in[r * 6 + j] = (iy >= 0 && iy < 28 && ix >= 0 && ix < 28)
                                    ? xb[iy * 28 + ix] : 0.0f;
            }
        }
        f16* hb = h1 + idx * H1_CH;       // NHWC8: 8 halves per position
        #pragma unroll
        for (int c = 0; c < 6; ++c) {
            const float* wc = w2eff + c * 36;
            float acc = 0.0f;
            #pragma unroll
            for (int i = 0; i < 36; ++i) acc = fmaf(in[i], wc[i], acc);
            hb[c] = (f16)acc;
            atomicAdd(&s[c], acc);
            atomicAdd(&s[6 + c], acc * acc);
        }
    }
    __syncthreads();
    if (tid < 12) atomicAdd(&stats[tid], s[tid]);
}

// BN scale/shift from accumulated sums. One block.
__global__ void k_bn_finalize(float* stats, int sumOff, int outOff, int C,
                              float invCount, const float* gamma, const float* beta) {
    int c = threadIdx.x;
    if (c >= C) return;
    float sum = stats[sumOff + c];
    float sq  = stats[sumOff + C + c];
    float mu  = sum * invCount;
    float var = sq * invCount - mu * mu;
    float rs  = rsqrtf(var + EPS);
    float g = gamma ? gamma[c] : 1.0f;
    float b = beta  ? beta[c]  : 0.0f;
    float scale = g * rs;
    float shift = b - mu * scale;
    stats[outOff + c]     = scale;
    stats[outOff + C + c] = shift;
}

// In-place BN + hardtanh(0,1) on f16 tensor with row stride; positions past
// `valid` are forced to zero (zeroes h1's NHWC8 channel pads every call).
__global__ void k_bn_apply(f16* h, const float* __restrict__ stats,
                           int scaleOff, int C, int perch, int stride, int valid,
                           long total) {
    long idx = (long)blockIdx.x * blockDim.x + threadIdx.x;
    if (idx >= total) return;
    int pos = (int)(idx % stride);
    float v = 0.0f;
    if (pos < valid) {
        int c = pos / perch;
        float sc = stats[scaleOff + c];
        float sh = stats[scaleOff + C + c];
        v = clip01((float)h[idx] * sc + sh);
    }
    h[idx] = (f16)v;
}

// conv2 (6->16, 5x5, valid) implicit GEMM via WMMA over NHWC8 input, fused
// 2x2 avgpool + bn2 sums. K-order k'=(ky*5+kx)*8+ch so every A-fragment chunk
// is one aligned b128 load. M rows ordered (pooledIdx*4+quad) so pooling
// collapses inside the C fragment. Each wave: 2 M tiles, 16 WMMAs.
__global__ void k_conv2_wmma(const f16* __restrict__ h1, const f16* __restrict__ wp,
                             f16* __restrict__ h2, float* __restrict__ stats) {
    __shared__ float s[32];
    int tid = threadIdx.x;
    if (tid < 32) s[tid] = 0.0f;
    __syncthreads();

    int lane = tid & 31;
    int wv   = tid >> 5;
    long gwp = (long)blockIdx.x * 4 + wv;        // 0 .. B*100/32 - 1
    int m  = lane & 15;
    int hs = lane >> 4;

    const f16* p[2];
    #pragma unroll
    for (int t = 0; t < 2; ++t) {
        long r   = gwp * 32 + t * 16 + m;
        long P   = r >> 2;
        int quad = (int)(r & 3);
        int b    = (int)(P / 25);
        int pp   = (int)(P % 25);
        int oy   = (pp / 5) * 2 + (quad >> 1);
        int ox   = (pp % 5) * 2 + (quad & 1);
        p[t] = h1 + (size_t)b * H1_STRIDE + (oy * 14 + ox) * 8;
    }

    v8f acc0 = {}, acc1 = {};
    #pragma unroll
    for (int ks = 0; ks < 8; ++ks) {
        int offA = conv2_chunk_off(ks * 4 + hs);
        int offB = conv2_chunk_off(ks * 4 + 2 + hs);
        union { v16h v; v8h h[2]; } u0, u1;
        u0.h[0] = *(const v8h*)(p[0] + offA);
        u0.h[1] = *(const v8h*)(p[0] + offB);
        u1.h[0] = *(const v8h*)(p[1] + offA);
        u1.h[1] = *(const v8h*)(p[1] + offB);
        v16h bf = *(const v16h*)(wp + (size_t)(ks * 32 + lane) * 16);
        acc0 = wmma_f16(u0.v, bf, acc0);
        acc1 = wmma_f16(u1.v, bf, acc1);
    }

    int ch = lane & 15;
    float tsum = 0.0f, tsq = 0.0f;
    #pragma unroll
    for (int t = 0; t < 2; ++t) {
        v8f& a = t ? acc1 : acc0;
        float v0 = (a[0] + a[1] + a[2] + a[3]) * 0.25f;
        float v1 = (a[4] + a[5] + a[6] + a[7]) * 0.25f;
        long Pb = gwp * 8 + t * 4 + hs * 2;
        {
            int bb = (int)(Pb / 25), pp = (int)(Pb % 25);
            h2[(size_t)bb * 400 + ch * 25 + pp] = (f16)v0;
        }
        {
            long P1 = Pb + 1;
            int bb = (int)(P1 / 25), pp = (int)(P1 % 25);
            h2[(size_t)bb * 400 + ch * 25 + pp] = (f16)v1;
        }
        tsum += v0 + v1;
        tsq  += v0 * v0 + v1 * v1;
    }
    atomicAdd(&s[ch], tsum);
    atomicAdd(&s[16 + ch], tsq);
    __syncthreads();
    if (tid < 32) atomicAdd(&stats[24 + tid], s[tid]);
}

// FC via WMMA with hardtanh(0,1) epilogue. A row-major [M, KDIM] f16, weights
// prepacked; out f16 [M, ldout]. Each wave: one M tile x two N tiles.
template <int KDIM, int KSTEPS, int NTILES>
__global__ void k_fc_wmma_clip(const f16* __restrict__ A, const f16* __restrict__ wp,
                               f16* __restrict__ out, int ldout) {
    constexpr int NPAIRS = NTILES / 2;
    int tid  = threadIdx.x;
    int lane = tid & 31;
    int wv   = tid >> 5;
    int gw = blockIdx.x * 4 + wv;
    int mt = gw / NPAIRS;
    int pr = gw % NPAIRS;
    int nt0 = 2 * pr;
    int m  = lane & 15;
    int hs = lane >> 4;
    const f16* arow = A + (size_t)(mt * 16 + m) * KDIM;

    v8f acc0 = {}, acc1 = {};
    #pragma unroll
    for (int ks = 0; ks < KSTEPS; ++ks) {
        const bool loValid = (ks * 32 + 16) <= KDIM;
        const bool hiValid = (ks * 32 + 32) <= KDIM;
        v8h lo{}, hi{};
        if (loValid) lo = *(const v8h*)(arow + ks * 32 + hs * 8);
        if (hiValid) hi = *(const v8h*)(arow + ks * 32 + 16 + hs * 8);
        union { v16h v; v8h h[2]; } u;
        u.h[0] = lo; u.h[1] = hi;
        v16h b0 = *(const v16h*)(wp + (size_t)((ks * NTILES + nt0) * 32 + lane) * 16);
        v16h b1 = *(const v16h*)(wp + (size_t)((ks * NTILES + nt0 + 1) * 32 + lane) * 16);
        acc0 = wmma_f16(u.v, b0, acc0);
        acc1 = wmma_f16(u.v, b1, acc1);
    }
    int col0 = nt0 * 16 + (lane & 15);
    #pragma unroll
    for (int v = 0; v < 8; ++v) {
        int row = mt * 16 + hs * 8 + v;
        out[(size_t)row * ldout + col0]      = (f16)clip01(acc0[v]);
        out[(size_t)row * ldout + col0 + 16] = (f16)clip01(acc1[v]);
    }
}

// FC3: 96 -> 16 (10 valid) via WMMA, f32 out [M,16], fused bn3 sums (cols<10).
__global__ void k_fc3_wmma_stats(const f16* __restrict__ A, const f16* __restrict__ wp,
                                 float* __restrict__ out, float* __restrict__ stats) {
    __shared__ float s[20];
    int tid = threadIdx.x;
    if (tid < 20) s[tid] = 0.0f;
    __syncthreads();

    int lane = tid & 31;
    int wv   = tid >> 5;
    int mt = blockIdx.x * 4 + wv;
    int m  = lane & 15;
    int hs = lane >> 4;
    const f16* arow = A + (size_t)(mt * 16 + m) * 96;

    v8f acc = {};
    #pragma unroll
    for (int ks = 0; ks < 3; ++ks) {
        v8h lo = *(const v8h*)(arow + ks * 32 + hs * 8);
        v8h hi = *(const v8h*)(arow + ks * 32 + 16 + hs * 8);
        union { v16h v; v8h h[2]; } u;
        u.h[0] = lo; u.h[1] = hi;
        v16h bf = *(const v16h*)(wp + (size_t)(ks * 32 + lane) * 16);
        acc = wmma_f16(u.v, bf, acc);
    }
    int col = lane & 15;
    float ls = 0.0f, lq = 0.0f;
    #pragma unroll
    for (int v = 0; v < 8; ++v) {
        int row = mt * 16 + hs * 8 + v;
        float val = acc[v];
        out[(size_t)row * 16 + col] = val;
        ls += val;
        lq += val * val;
    }
    if (col < 10) {
        atomicAdd(&s[col], ls);
        atomicAdd(&s[10 + col], lq);
    }
    __syncthreads();
    if (tid < 20) atomicAdd(&stats[88 + tid], s[tid]);
}

// Final 1D batchnorm (affine=False) on [B,10] from h5 [B,16].
__global__ void k_final_bn(const float* __restrict__ h5, const float* __restrict__ stats,
                           float* __restrict__ out) {
    long idx = (long)blockIdx.x * blockDim.x + threadIdx.x;
    const long total = (long)BATCH * 10;
    if (idx >= total) return;
    int b = (int)(idx / 10);
    int j = (int)(idx % 10);
    out[idx] = h5[(size_t)b * 16 + j] * stats[108 + j] + stats[118 + j];
}

// ---------------------------------------------------------------------------

extern "C" void kernel_launch(void* const* d_in, const int* in_sizes, int n_in,
                              void* d_out, int out_size, void* d_ws, size_t ws_size,
                              hipStream_t stream) {
    const float* x     = (const float*)d_in[0];   // [B,1,28,28]
    const float* w1    = (const float*)d_in[1];   // [6,1,5,5]
    const float* w2    = (const float*)d_in[2];   // [16,6,5,5]
    const float* bn1_g = (const float*)d_in[3];   // [6]
    const float* bn1_b = (const float*)d_in[4];
    const float* bn2_g = (const float*)d_in[5];   // [16]
    const float* bn2_b = (const float*)d_in[6];
    const float* fw1   = (const float*)d_in[7];   // [120,400]
    const float* fw2   = (const float*)d_in[8];   // [84,120]
    const float* fw3   = (const float*)d_in[9];   // [10,84]
    float* out = (float*)d_out;                   // [B,10]

    const long B = BATCH;
    char* ws = (char*)d_ws;
    size_t off = 0;
    auto carve = [&](size_t bytes) -> char* {
        char* p = ws + off;
        off = (off + bytes + 255) & ~(size_t)255;
        return p;
    };
    f16*   h1    = (f16*)  carve(B * H1_STRIDE * sizeof(f16)); // conv1 out, NHWC8
    f16*   h2    = (f16*)  carve(B * 400 * sizeof(f16));       // conv2 pooled / FC1 in
    f16*   h3    = (f16*)  carve(B * 128 * sizeof(f16));       // FC1 out (padded)
    f16*   h4    = (f16*)  carve(B * 96  * sizeof(f16));       // FC2 out (padded)
    float* h5    = (float*)carve(B * 16  * sizeof(float));     // FC3 out (padded)
    f16*   wp2   = (f16*)  carve(8  * 1 * 512 * sizeof(f16));  // conv2 pack K256 N16
    f16*   wpf1  = (f16*)  carve(13 * 8 * 512 * sizeof(f16));  // fc1 pack K416 N128
    f16*   wpf2  = (f16*)  carve(4  * 6 * 512 * sizeof(f16));  // fc2 pack K128 N96
    f16*   wpf3  = (f16*)  carve(3  * 1 * 512 * sizeof(f16));  // fc3 pack K96  N16
    float* w2eff = (float*)carve(216 * sizeof(float));         // conv1 6x6 eff filter
    float* stats = (float*)carve(128 * sizeof(float));
    (void)ws_size; (void)in_sizes; (void)n_in; (void)out_size;

    k_zero_stats<<<1, 128, 0, stream>>>(stats);

    // prep (deterministic, cheap)
    k_prep_pack_conv2<<<(8 * 512 + 255) / 256, 256, 0, stream>>>(w2, wp2);
    k_prep_pack<<<(13 * 8 * 512 + 255) / 256, 256, 0, stream>>>(fw1, wpf1, 400, 120, 13, 8);
    k_prep_pack<<<(4 * 6 * 512 + 255) / 256, 256, 0, stream>>>(fw2, wpf2, 120, 84,  4, 6);
    k_prep_pack<<<(3 * 1 * 512 + 255) / 256, 256, 0, stream>>>(fw3, wpf3, 84,  10,  3, 1);
    k_prep_w2eff<<<1, 256, 0, stream>>>(w1, w2eff);

    // conv1 + pool (6x6 stride-2 form) + bn1 stats
    long t0 = B * H1_POS;
    k_conv1_pool<<<(unsigned)((t0 + 255) / 256), 256, 0, stream>>>(x, w2eff, h1, stats);
    k_bn_finalize<<<1, 32, 0, stream>>>(stats, 0, 12, 6, 1.0f / (float)(B * 196), bn1_g, bn1_b);
    long t1 = B * H1_STRIDE;
    k_bn_apply<<<(unsigned)((t1 + 255) / 256), 256, 0, stream>>>(
        h1, stats, 12, 6, /*perch=*/1, /*stride=*/H1_CH, /*valid=*/6, t1);

    // conv2 (WMMA, 2 M tiles/wave, aligned b128 gather) + pool + bn2 stats
    k_conv2_wmma<<<(unsigned)(B * 100 / 128), 128, 0, stream>>>(h1, wp2, h2, stats);
    k_bn_finalize<<<1, 32, 0, stream>>>(stats, 24, 56, 16, 1.0f / (float)(B * 25), bn2_g, bn2_b);
    long t2 = B * 400;
    k_bn_apply<<<(unsigned)((t2 + 255) / 256), 256, 0, stream>>>(
        h2, stats, 56, 16, /*perch=*/25, /*stride=*/400, /*valid=*/400, t2);

    // FC chain (WMMA, 2 N tiles/wave)
    int mtiles = (int)(B / 16);                                   // 1024
    k_fc_wmma_clip<400, 13, 8><<<(unsigned)(mtiles * 4 / 4), 128, 0, stream>>>(h2, wpf1, h3, 128);
    k_fc_wmma_clip<128, 4, 6><<<(unsigned)(mtiles * 3 / 4), 128, 0, stream>>>(h3, wpf2, h4, 96);
    k_fc3_wmma_stats<<<(unsigned)(mtiles / 4), 128, 0, stream>>>(h4, wpf3, h5, stats);

    // final 1d batchnorm (affine=False)
    k_bn_finalize<<<1, 32, 0, stream>>>(stats, 88, 108, 10, 1.0f / (float)B, nullptr, nullptr);
    k_final_bn<<<(unsigned)((B * 10 + 255) / 256), 256, 0, stream>>>(h5, stats, out);
}
